// PA_Module_4475355922661
// MI455X (gfx1250) — matrix operand order repaired and checked by reading, not compiled, not run on hardware
//
#include <hip/hip_runtime.h>
#include <math.h>

// MI455X (gfx1250) fused channel-attention kernel, round 3.
// - 32 i-rows per workgroup: each B fragment (x_fix, f32->bf16 inline) feeds
//   TWO v_wmma_f32_16x16x32_bf16 ops -> halves B-side load+convert per wmma.
// - A strips (x_ref rows) staged in LDS as bf16 per K-chunk, shared by waves.
// - Energy tiles stored + softmax stats computed one 16-row half at a time,
//   reusing a single 32 KB LDS strip (total static LDS ~50 KB).

typedef __attribute__((ext_vector_type(16))) __bf16 v16bf;
typedef __attribute__((ext_vector_type(8)))  __bf16 v8bf;
typedef __attribute__((ext_vector_type(8)))  float  v8f;

#define BATCH 8
#define CCH   512
#define SSP   4096   // 64*64 spatial
#define KCH   256    // K-chunk staged in LDS (32 rows x 256 bf16 = 16 KB)

// Convert 16 contiguous f32 -> v16bf (B fragment: 16 contiguous K values)
__device__ __forceinline__ v16bf cvt16(const float* __restrict__ p) {
  const float4* q = (const float4*)p;
  float4 f0 = q[0], f1 = q[1], f2 = q[2], f3 = q[3];
  v16bf v;
  v[0]=(__bf16)f0.x;  v[1]=(__bf16)f0.y;  v[2]=(__bf16)f0.z;  v[3]=(__bf16)f0.w;
  v[4]=(__bf16)f1.x;  v[5]=(__bf16)f1.y;  v[6]=(__bf16)f1.z;  v[7]=(__bf16)f1.w;
  v[8]=(__bf16)f2.x;  v[9]=(__bf16)f2.y;  v[10]=(__bf16)f2.z; v[11]=(__bf16)f2.w;
  v[12]=(__bf16)f3.x; v[13]=(__bf16)f3.y; v[14]=(__bf16)f3.z; v[15]=(__bf16)f3.w;
  return v;
}

__device__ __forceinline__ v8bf pack8(float4 a, float4 b) {
  v8bf v;
  v[0]=(__bf16)a.x; v[1]=(__bf16)a.y; v[2]=(__bf16)a.z; v[3]=(__bf16)a.w;
  v[4]=(__bf16)b.x; v[5]=(__bf16)b.y; v[6]=(__bf16)b.z; v[7]=(__bf16)b.w;
  return v;
}

// A fragment from LDS: lane holds K = g*16..+15 (contiguous, matches B)
__device__ __forceinline__ v16bf afrag_lds(const __bf16* __restrict__ aBase,
                                           int k0, int g) {
  v8bf lo = *(const v8bf*)(aBase + k0 + g * 16);
  v8bf hi = *(const v8bf*)(aBase + k0 + g * 16 + 8);
  return __builtin_shufflevector(lo, hi, 0, 1, 2, 3, 4, 5, 6, 7,
                                 8, 9, 10, 11, 12, 13, 14, 15);
}

__global__ __launch_bounds__(256)
void pa_fused_kernel(const float* __restrict__ x_fix,
                     const float* __restrict__ x_ref,
                     const float* __restrict__ gptr,
                     float* __restrict__ out) {
  __shared__ __attribute__((aligned(16))) __bf16 aLDS[32 * KCH]; // 16 KB
  __shared__ float eLDS[16 * CCH];                               // 32 KB strip
  __shared__ float red[256];
  __shared__ float rowmin[16];
  __shared__ float rowZ[16];
  __shared__ float rowM[32];

  const int tid  = threadIdx.x;
  const int lane = tid & 31;
  const int wave = tid >> 5;          // 8 waves
  const int g    = lane >> 4;         // lane-group (0/1)
  const int ln16 = lane & 15;
  const int b    = blockIdx.y;        // batch
  const int i0   = blockIdx.x * 32;   // 32-row i-strip base

  const float gamma = gptr[0];
  const size_t batchBase = (size_t)b * CCH * SSP;

  // ---------------- Phase 1: energy via bf16 WMMA ----------------
  // Per-wave j-tiles: jt = wave + 8*t. B operand lane: x_fix row jt*16+ln16,
  // lane group selects K half (16 contiguous K).
  const float* __restrict__ fixRow0 =
      x_fix + batchBase + (size_t)(wave * 16 + ln16) * SSP + g * 16;

  // Staging: 32 rows x 256 K per chunk; thread covers row tid>>3, 32 K values
  const int sRow = tid >> 3;
  const int sK   = (tid & 7) * 32;
  const float* __restrict__ stageSrc =
      x_ref + batchBase + (size_t)(i0 + sRow) * SSP + sK;
  v8bf* __restrict__ stageDst = (v8bf*)(aLDS + sRow * KCH + sK);

  v8f acc[8];   // [strip*4 + t]; fully-unrolled constant indexing only
#pragma unroll
  for (int i = 0; i < 8; ++i) acc[i] = (v8f){};

  const __bf16* __restrict__ aBase0 = aLDS + ln16 * KCH;          // rows 0..15
  const __bf16* __restrict__ aBase1 = aLDS + (16 + ln16) * KCH;   // rows 16..31

  for (int kc = 0; kc < SSP; kc += KCH) {
    __syncthreads();   // all waves done reading previous A chunk
    {
      const float4* q = (const float4*)(stageSrc + kc);
#pragma unroll
      for (int i = 0; i < 4; ++i)
        stageDst[i] = pack8(q[2 * i], q[2 * i + 1]);
    }
    __syncthreads();   // A chunk visible to all waves

#pragma unroll
    for (int t = 0; t < 4; ++t) {
      const float* __restrict__ fixRow = fixRow0 + (size_t)(t * 8 * 16) * SSP + kc;
#pragma unroll 2
      for (int k0 = 0; k0 < KCH; k0 += 32) {
        __builtin_prefetch(fixRow + k0 + 512, 0, 0);      // global_prefetch_b8
        v16bf bfrag = cvt16(fixRow + k0);                 // shared by 2 wmmas
        v16bf a0 = afrag_lds(aBase0, k0, g);
        v16bf a1 = afrag_lds(aBase1, k0, g);
        acc[t] = __builtin_amdgcn_wmma_f32_16x16x32_bf16(
            false, a0, false, bfrag, (short)0, acc[t], false, false);
        acc[4 + t] = __builtin_amdgcn_wmma_f32_16x16x32_bf16(
            false, a1, false, bfrag, (short)0, acc[4 + t], false, false);
      }
    }
  }

  // ---------------- Phase 2: tiles -> LDS + softmax stats, per 16-row half ---
  // D layout: VGPR r holds (M = r + 8*g, N = ln16).
  // attention = softmax(rowmax - e) ==> stable form uses exp(rowmin - e).
  const int row = tid >> 4;           // 16 rows, 16 threads each
  const int sub = tid & 15;           // each covers 32 columns
  const float* __restrict__ e = eLDS + row * CCH + sub * 32;

#pragma unroll
  for (int half = 0; half < 2; ++half) {
    __syncthreads();   // previous half's stats reads complete
#pragma unroll
    for (int t = 0; t < 4; ++t) {
      int jt = wave + t * 8;
#pragma unroll
      for (int r = 0; r < 8; ++r) {
        int M = r + (g << 3);
        eLDS[M * CCH + jt * 16 + ln16] = acc[half * 4 + t][r];
      }
    }
    __syncthreads();

    float mn = INFINITY;
#pragma unroll 8
    for (int c = 0; c < 32; ++c) mn = fminf(mn, e[c]);
    red[tid] = mn;
    __syncthreads();
    if (sub == 0) {
      float m2 = red[tid];
#pragma unroll
      for (int k = 1; k < 16; ++k) m2 = fminf(m2, red[tid + k]);
      rowmin[row] = m2;
    }
    __syncthreads();

    const float rm = rowmin[row];
    float z = 0.f;
#pragma unroll 8
    for (int c = 0; c < 32; ++c) z += __expf(rm - e[c]);
    red[tid] = z;
    __syncthreads();
    if (sub == 0) {
      float zs = 0.f;
#pragma unroll
      for (int k = 0; k < 16; ++k) zs += red[tid + k];
      rowZ[row] = zs;
    }
    __syncthreads();

    // faithful recompute of sum_j attention_j (== ~1 numerically)
    const float zr = rowZ[row];
    float asum = 0.f;
#pragma unroll 8
    for (int c = 0; c < 32; ++c) asum += __expf(rm - e[c]) / zr;
    red[tid] = asum;
    __syncthreads();
    if (sub == 0) {
      float as = 0.f;
#pragma unroll
      for (int k = 0; k < 16; ++k) as += red[tid + k];
      float sig = 1.f / (1.f + __expf(-as));        // channel scale
      rowM[half * 16 + row] = 1.f + gamma * sig;    // out = ref*(1+gamma*sig)
    }
  }
  __syncthreads();

  // ---------------- Phase 3: scaled residual output ----------------
  const size_t stripBase = batchBase + (size_t)i0 * SSP;
  const float4* __restrict__ refV = (const float4*)(x_ref + stripBase);
  float4* __restrict__ outV = (float4*)(out + stripBase);
#pragma unroll 4
  for (int it = 0; it < 128; ++it) {
    int lin = it * 256 + tid;         // 32768 float4 = 32 rows * 4096 f32
    int r = lin >> 10;                // 1024 float4 per row
    float m = rowM[r];
    float4 v = refV[lin];
    v.x *= m; v.y *= m; v.z *= m; v.w *= m;
    outV[lin] = v;
  }
}

extern "C" void kernel_launch(void* const* d_in, const int* in_sizes, int n_in,
                              void* d_out, int out_size, void* d_ws, size_t ws_size,
                              hipStream_t stream) {
  const float* x_fix = (const float*)d_in[0];
  const float* x_ref = (const float*)d_in[1];
  const float* gamma = (const float*)d_in[2];
  float* out = (float*)d_out;
  (void)in_sizes; (void)n_in; (void)out_size; (void)d_ws; (void)ws_size;

  dim3 grid(CCH / 32, BATCH);   // 16 i-strips x 8 batches
  dim3 block(256);              // 8 wave32
  pa_fused_kernel<<<grid, block, 0, stream>>>(x_fix, x_ref, gamma, out);
}
